// TeaBlock_61186104099423
// MI455X (gfx1250) — matrix operand order repaired and checked by reading, not compiled, not run on hardware
//
#include <hip/hip_runtime.h>
#include <hip/hip_bf16.h>

// ---------------------------------------------------------------------------
// MI455X (gfx1250) fused "TeaBlock" pipeline, bf16 WMMA (f32 accumulate).
//   prep:    transpose+convert weights to bf16 (N-major, zero-padded)
//   layer1:  15 window GEMMs + tail GEMM  -> Y (B,1024) bf16   (in ws, L2-resident)
//   layer23: group GEMMs + final GEMM + pool + softmax -> out (B,10) f32
// ---------------------------------------------------------------------------

typedef __attribute__((ext_vector_type(16))) __bf16 v16bf;
typedef __attribute__((ext_vector_type(8)))  float  v8f;

union FragBF { uint4 q[2]; v16bf v; };   // 32 bytes = one 16x32 bf16 A/B fragment per lane

#define B_TOTAL 16384
#define XDIM    2176

// workspace layout (bytes)
#define OFF_W1T   0u            // [15][64][256] bf16
#define OFF_W16T  491520u       // [64][384]     bf16 (K padded 356->384)
#define OFF_W2T   540672u       // [4][64][256]  bf16
#define OFF_W3T   671744u       // [272][256]    bf16 (N padded 260->272)
#define OFF_Y     811008u       // [16384][1024] bf16

// ---------------------------------------------------------------------------
__global__ void prep_weights(const float* __restrict__ W1, const float* __restrict__ W16,
                             const float* __restrict__ W2, const float* __restrict__ W3,
                             __bf16* __restrict__ W1T, __bf16* __restrict__ W16T,
                             __bf16* __restrict__ W2T, __bf16* __restrict__ W3T) {
    int idx = blockIdx.x * blockDim.x + threadIdx.x;
    int stride = gridDim.x * blockDim.x;
    // W1T[h][n][k] = W1[h][k][n]
    for (int i = idx; i < 15 * 64 * 256; i += stride) {
        int h = i / (64 * 256), rem = i % (64 * 256);
        int n = rem >> 8, k = rem & 255;
        W1T[i] = (__bf16)W1[(h * 256 + k) * 64 + n];
    }
    // W16T[n][k] = W16[k][n], K padded to 384
    for (int i = idx; i < 64 * 384; i += stride) {
        int n = i / 384, k = i % 384;
        W16T[i] = (__bf16)(k < 356 ? W16[k * 64 + n] : 0.0f);
    }
    // W2T[j][n][k] = W2[j][k][n]
    for (int i = idx; i < 4 * 64 * 256; i += stride) {
        int j = i / (64 * 256), rem = i % (64 * 256);
        int n = rem >> 8, k = rem & 255;
        W2T[i] = (__bf16)W2[(j * 256 + k) * 64 + n];
    }
    // W3T[n][k] = W3[k][n], N padded 260 -> 272 with zeros
    for (int i = idx; i < 272 * 256; i += stride) {
        int n = i >> 8, k = i & 255;
        W3T[i] = (__bf16)(n < 260 ? W3[k * 260 + n] : 0.0f);
    }
}

// ---------------------------------------------------------------------------
// Layer-1: one workgroup = 128 batch rows x one head (0..14 windows, 15 tail).
// 8 waves, wave w owns M-tile w (16 rows) x all 4 N-tiles.
template <int KC, int KVALID>
__device__ __forceinline__ void layer1_body(const float* __restrict__ x,
                                            const __bf16* __restrict__ Wt,
                                            const float* __restrict__ bias,
                                            __bf16* __restrict__ Y,
                                            __bf16* lds, int rowBase, int colStart,
                                            int headColBase, int tid) {
    constexpr int SA = KC + 8;            // bf16 stride pad: conflict-free ds_load_b128
    constexpr int KSTEPS = KC / 32;
    __bf16* As = lds;                     // [128][SA]
    __bf16* Bs = lds + 128 * SA;          // [64][SA]

    const int lane = tid & 31;
    const int wave = tid >> 5;            // 0..7
    const int lanelo = lane & 15;
    const int ksel = lane >> 4;           // 0: lanes 0-15, 1: lanes 16-31

    // stage X tile (f32 -> bf16) into LDS
    for (int i = tid; i < 128 * KC; i += 256) {
        int r = i / KC, c = i % KC;
        float v = (c < KVALID) ? x[(size_t)(rowBase + r) * XDIM + colStart + c] : 0.0f;
        As[r * SA + c] = (__bf16)v;
    }
    // stage W^T tile (already bf16, N-major contiguous K)
    for (int i = tid; i < 64 * KC; i += 256) {
        int r = i / KC, c = i % KC;
        Bs[r * SA + c] = Wt[i];
    }
    __syncthreads();

    v8f acc[4] = {};
    const __bf16* arow = As + (wave * 16 + lanelo) * SA;
#pragma unroll
    for (int k = 0; k < KSTEPS; ++k) {
        FragBF fa;
        const __bf16* ap = arow + k * 32 + ksel * 8;   // A: K-chunks per ISA layout
        fa.q[0] = *(const uint4*)(ap);
        fa.q[1] = *(const uint4*)(ap + 16);
#pragma unroll
        for (int n = 0; n < 4; ++n) {
            FragBF fb;
            const __bf16* bp = Bs + (n * 16 + lanelo) * SA + k * 32 + ksel * 8;
            fb.q[0] = *(const uint4*)(bp);
            fb.q[1] = *(const uint4*)(bp + 16);
            acc[n] = __builtin_amdgcn_wmma_f32_16x16x32_bf16(
                false, fa.v, false, fb.v, (short)0, acc[n], false, false);
        }
    }

    const int mrow0 = rowBase + wave * 16 + ksel * 8;  // C/D: lanes 16-31 hold M = r+8
#pragma unroll
    for (int n = 0; n < 4; ++n) {
        float bv = bias[n * 16 + lanelo];
        int col = headColBase + n * 16 + lanelo;
#pragma unroll
        for (int r = 0; r < 8; ++r) {
            float v = acc[n][r] + bv;
            v = v > 0.0f ? v : 0.0f;
            Y[(size_t)(mrow0 + r) * 1024 + col] = (__bf16)v;
        }
    }
}

__global__ void __launch_bounds__(256, 1)
layer1_kernel(const float* __restrict__ x, const __bf16* __restrict__ W1T,
              const float* __restrict__ b1, const __bf16* __restrict__ W16T,
              const float* __restrict__ b16, __bf16* __restrict__ Y) {
    extern __shared__ __bf16 lds1[];
    const int tile = blockIdx.x;          // 0..127  (128 rows each)
    const int head = blockIdx.y;          // 0..15
    const int tid = threadIdx.x;
    const int rowBase = tile * 128;
    if (head == 15) {
        layer1_body<384, 356>(x, W16T, b16, Y, lds1, rowBase, 1820, 15 * 64, tid);
    } else {
        layer1_body<256, 256>(x, W1T + head * 64 * 256, b1 + head * 64, Y, lds1,
                              rowBase, head * 128, head * 64, tid);
    }
}

// ---------------------------------------------------------------------------
// Layer-2 + Layer-3 + pool + softmax: one workgroup = 64 batch rows, 4 waves.
#define ZS_STRIDE 264
#define HS_STRIDE 280

__global__ void __launch_bounds__(128, 1)
layer23_kernel(const __bf16* __restrict__ Y, const __bf16* __restrict__ W2T,
               const float* __restrict__ b2, const __bf16* __restrict__ W3T,
               const float* __restrict__ b3, float* __restrict__ out) {
    extern __shared__ unsigned char lds2[];
    __bf16* zs = (__bf16*)lds2;                                    // [64][264] bf16
    float*  hs = (float*)(lds2 + 64 * ZS_STRIDE * 2);              // [64][280] f32

    const int tile = blockIdx.x;          // 0..255 (64 rows each)
    const int tid = threadIdx.x;
    const int lane = tid & 31;
    const int wave = tid >> 5;            // 0..3
    const int lanelo = lane & 15;
    const int ksel = lane >> 4;
    const int rowBase = tile * 64;
    const int mbase = wave * 16;

    // ---- layer 2: z = relu(g @ W2 + b2), A-frags straight from Y (L2-resident)
    const __bf16* arowY = Y + (size_t)(rowBase + mbase + lanelo) * 1024;
#pragma unroll
    for (int j = 0; j < 4; ++j) {
#pragma unroll
        for (int n = 0; n < 4; ++n) {
            v8f acc = {};
#pragma unroll
            for (int k = 0; k < 8; ++k) {
                FragBF fa, fb;
                const __bf16* ap = arowY + j * 256 + k * 32 + ksel * 8;
                fa.q[0] = *(const uint4*)(ap);
                fa.q[1] = *(const uint4*)(ap + 16);
                const __bf16* bp = W2T + (j * 64 + n * 16 + lanelo) * 256 + k * 32 + ksel * 8;
                fb.q[0] = *(const uint4*)(bp);
                fb.q[1] = *(const uint4*)(bp + 16);
                acc = __builtin_amdgcn_wmma_f32_16x16x32_bf16(
                    false, fa.v, false, fb.v, (short)0, acc, false, false);
            }
            float bv = b2[j * 64 + n * 16 + lanelo];
            int zcol = j * 64 + n * 16 + lanelo;
#pragma unroll
            for (int r = 0; r < 8; ++r) {
                float v = acc[r] + bv;
                v = v > 0.0f ? v : 0.0f;
                zs[(mbase + ksel * 8 + r) * ZS_STRIDE + zcol] = (__bf16)v;
            }
        }
    }
    __syncthreads();

    // ---- layer 3: h = relu(z @ W3 + b3), N padded to 272 (17 tiles)
    const __bf16* zrow = zs + (mbase + lanelo) * ZS_STRIDE;
    for (int n = 0; n < 17; ++n) {
        v8f acc = {};
#pragma unroll
        for (int k = 0; k < 8; ++k) {
            FragBF fa, fb;
            const __bf16* ap = zrow + k * 32 + ksel * 8;
            fa.q[0] = *(const uint4*)(ap);
            fa.q[1] = *(const uint4*)(ap + 16);
            const __bf16* bp = W3T + (n * 16 + lanelo) * 256 + k * 32 + ksel * 8;
            fb.q[0] = *(const uint4*)(bp);
            fb.q[1] = *(const uint4*)(bp + 16);
            acc = __builtin_amdgcn_wmma_f32_16x16x32_bf16(
                false, fa.v, false, fb.v, (short)0, acc, false, false);
        }
        int col = n * 16 + lanelo;
        float bv = (col < 260) ? b3[col] : 0.0f;
#pragma unroll
        for (int r = 0; r < 8; ++r) {
            float v = acc[r] + bv;
            v = v > 0.0f ? v : 0.0f;
            hs[(mbase + ksel * 8 + r) * HS_STRIDE + col] = v;
        }
    }
    __syncthreads();

    // ---- pool (10 groups of 26) + softmax, one thread per batch row
    if (tid < 64) {
        const float* hrow = hs + tid * HS_STRIDE;
        float p[10];
#pragma unroll
        for (int c = 0; c < 10; ++c) {
            float s = 0.0f;
            for (int i = 0; i < 26; ++i) s += hrow[c * 26 + i];
            p[c] = s;
        }
        float mx = p[0];
#pragma unroll
        for (int c = 1; c < 10; ++c) mx = fmaxf(mx, p[c]);
        float den = 0.0f;
#pragma unroll
        for (int c = 0; c < 10; ++c) { p[c] = __expf(p[c] - mx); den += p[c]; }
        float inv = 1.0f / den;
        float* orow = out + (size_t)(rowBase + tid) * 10;
#pragma unroll
        for (int c = 0; c < 10; ++c) orow[c] = p[c] * inv;
    }
}

// ---------------------------------------------------------------------------
extern "C" void kernel_launch(void* const* d_in, const int* in_sizes, int n_in,
                              void* d_out, int out_size, void* d_ws, size_t ws_size,
                              hipStream_t stream) {
    const float* x   = (const float*)d_in[0];
    const float* W1  = (const float*)d_in[1];
    const float* b1  = (const float*)d_in[2];
    const float* W16 = (const float*)d_in[3];
    const float* b16 = (const float*)d_in[4];
    const float* W2  = (const float*)d_in[5];
    const float* b2  = (const float*)d_in[6];
    const float* W3  = (const float*)d_in[7];
    const float* b3  = (const float*)d_in[8];
    float* out = (float*)d_out;

    char* ws = (char*)d_ws;
    __bf16* W1T  = (__bf16*)(ws + OFF_W1T);
    __bf16* W16T = (__bf16*)(ws + OFF_W16T);
    __bf16* W2T  = (__bf16*)(ws + OFF_W2T);
    __bf16* W3T  = (__bf16*)(ws + OFF_W3T);
    __bf16* Yb   = (__bf16*)(ws + OFF_Y);

    prep_weights<<<512, 256, 0, stream>>>(W1, W16, W2, W3, W1T, W16T, W2T, W3T);

    dim3 g1(B_TOTAL / 128, 16);
    size_t lds1 = (size_t)(128 + 64) * (384 + 8) * sizeof(__bf16);   // worst case (head 15)
    layer1_kernel<<<g1, 256, lds1, stream>>>(x, W1T, b1, W16T, b16, Yb);

    size_t lds2 = (size_t)64 * ZS_STRIDE * 2 + (size_t)64 * HS_STRIDE * 4;
    layer23_kernel<<<B_TOTAL / 64, 128, lds2, stream>>>(Yb, W2T, b2, W3T, b3, out);
}
